// MultiHeadAttention_with_RoPE_33732673143628
// MI455X (gfx1250) — compile-verified, matrix operand order checked by reading, not run on hardware
//
#include <hip/hip_runtime.h>
#include <hip/hip_bf16.h>

typedef __bf16 bf16_t;
typedef __attribute__((ext_vector_type(16))) __bf16 v16bf;
typedef __attribute__((ext_vector_type(8)))  float  v8f;

union FragU { v16bf v; uint4 q[2]; };
union Pack8 { bf16_t h[8]; uint4 q; };

// D = A(16x32) x B(32x16) + C ; bf16 in, fp32 out
__device__ __forceinline__ v8f wmma_bf16(v16bf a, v16bf b, v8f c) {
  return __builtin_amdgcn_wmma_f32_16x16x32_bf16(false, a, false, b, (short)0, c, false, false);
}

// Load a 16x32 (rows x k) bf16 fragment from a row-major buffer.
// ISA layout: lanes 0-15 hold row=lane with k in {0..7, 16..23};
// lanes 16-31 hold row=lane-16 with k in {8..15, 24..31}.
__device__ __forceinline__ v16bf load_frag_rm(const bf16_t* base, int ld) {
  const int lane = threadIdx.x & 31;
  const int r  = lane & 15;
  const int kb = (lane >> 4) << 3;           // 0 or 8
  const bf16_t* p = base + (long)r * ld + kb;
  FragU f;
  f.q[0] = *(const uint4*)(p);               // k = kb .. kb+7
  f.q[1] = *(const uint4*)(p + 16);          // k = kb+16 .. kb+23
  return f.v;
}

// CDNA5 async global->LDS copy, 16B per lane, tracked by ASYNCcnt.
__device__ __forceinline__ void async_copy_b128(void* lds_dst, const void* gsrc) {
  unsigned loff = (unsigned)(unsigned long long)(size_t)lds_dst;  // LDS byte offset
  asm volatile("global_load_async_to_lds_b128 %0, %1, off"
               :: "v"(loff), "v"(gsrc) : "memory");
}
__device__ __forceinline__ void wait_async_all() {
  asm volatile("s_wait_asynccnt 0" ::: "memory");
}

// ---- fused DPP butterfly reductions: one VALU instruction per stage --------
// dpp8 on src0: t = op(t^mask, t); row_ror:8 on 16-lane rows == lane^8.
__device__ __forceinline__ float rowmax16(float t) {
  float d;
  asm("v_max_num_f32_dpp %0, %1, %1 dpp8:[1,0,3,2,5,4,7,6]" : "=v"(d) : "v"(t));
  asm("v_max_num_f32_dpp %0, %1, %1 dpp8:[2,3,0,1,6,7,4,5]" : "=v"(t) : "v"(d));
  asm("v_max_num_f32_dpp %0, %1, %1 dpp8:[4,5,6,7,0,1,2,3]" : "=v"(d) : "v"(t));
  asm("v_max_num_f32_dpp %0, %1, %1 row_ror:8 row_mask:0xf bank_mask:0xf"
      : "=v"(t) : "v"(d));
  return t;
}
__device__ __forceinline__ float rowsum16(float t) {
  float d;
  asm("v_add_f32_dpp %0, %1, %1 dpp8:[1,0,3,2,5,4,7,6]" : "=v"(d) : "v"(t));
  asm("v_add_f32_dpp %0, %1, %1 dpp8:[2,3,0,1,6,7,4,5]" : "=v"(t) : "v"(d));
  asm("v_add_f32_dpp %0, %1, %1 dpp8:[4,5,6,7,0,1,2,3]" : "=v"(d) : "v"(t));
  asm("v_add_f32_dpp %0, %1, %1 row_ror:8 row_mask:0xf bank_mask:0xf"
      : "=v"(t) : "v"(d));
  return t;
}
// raw max without clang's canonicalize self-maxes
__device__ __forceinline__ float fmax_raw(float a, float b) {
  float d;
  asm("v_max_num_f32 %0, %1, %2" : "=v"(d) : "v"(a), "v"(b));
  return d;
}

// -------------------------------------------------------------------------
// fp32 -> bf16 elementwise convert
__global__ void f32_to_bf16_kernel(const float* __restrict__ in,
                                   bf16_t* __restrict__ out, long n) {
  long i = (long)blockIdx.x * blockDim.x + threadIdx.x;
  if (i < n) out[i] = (bf16_t)in[i];
}

// -------------------------------------------------------------------------
// NT GEMM: C[M,N] = A[M,K] * B[N,K]^T  (A,B bf16 row-major)
// 128x128 block tile, 256 threads (8 waves, 2x4), wave tile 64x32, K-step 32.
// Async double-buffered LDS staging (ASYNCcnt pipeline).
// OUTMODE: 0 = fp32 row-major, 1 = bf16 row-major, 2 = bf16 transposed
//          (Cout[(b*N + col)*2048 + s] with row = b*2048 + s) for the V path.
template<int OUTMODE>
__global__ __launch_bounds__(256) void gemm_nt_kernel(
    const bf16_t* __restrict__ A, const bf16_t* __restrict__ B,
    void* __restrict__ Cout, int M, int N, int K)
{
  __shared__ bf16_t As[2][128][40];   // padded stride: 80B rows, 16B aligned
  __shared__ bf16_t Bs[2][128][40];

  const int t     = threadIdx.x;
  const int lane  = t & 31;
  const int wid   = t >> 5;
  const int waveM = wid & 1;       // 2 waves along M (64 rows each)
  const int waveN = wid >> 1;      // 4 waves along N (32 cols each)
  const long blockM = (long)blockIdx.y * 128;
  const int  blockN = blockIdx.x * 128;

  v8f acc[4][2];
  const v8f vzero = {0.f,0.f,0.f,0.f,0.f,0.f,0.f,0.f};
#pragma unroll
  for (int mt = 0; mt < 4; ++mt)
#pragma unroll
    for (int nt = 0; nt < 2; ++nt) acc[mt][nt] = vzero;

  const int ldRow  = t >> 1;       // 128 rows, 2 threads/row
  const int ldHalf = (t & 1) * 16; // 16 bf16 = 32B each

  auto stage = [&](int buf, int k0) {
    const bf16_t* ag = A + (blockM + ldRow) * K + k0 + ldHalf;
    const bf16_t* bg = B + (long)(blockN + ldRow) * K + k0 + ldHalf;
    async_copy_b128(&As[buf][ldRow][ldHalf],     ag);
    async_copy_b128(&As[buf][ldRow][ldHalf + 8], ag + 8);
    async_copy_b128(&Bs[buf][ldRow][ldHalf],     bg);
    async_copy_b128(&Bs[buf][ldRow][ldHalf + 8], bg + 8);
  };

  stage(0, 0);
  int cur = 0;
  for (int k0 = 0; k0 < K; k0 += 32) {
    wait_async_all();              // previous stage landed in LDS (this wave)
    __syncthreads();               // ... and in every other wave
    if (k0 + 32 < K) stage(cur ^ 1, k0 + 32);   // async fill of other buffer

    v16bf bf0 = load_frag_rm(&Bs[cur][waveN * 32][0], 40);
    v16bf bf1 = load_frag_rm(&Bs[cur][waveN * 32 + 16][0], 40);
#pragma unroll
    for (int mt = 0; mt < 4; ++mt) {
      v16bf af = load_frag_rm(&As[cur][waveM * 64 + mt * 16][0], 40);
      acc[mt][0] = wmma_bf16(af, bf0, acc[mt][0]);
      acc[mt][1] = wmma_bf16(af, bf1, acc[mt][1]);
    }
    __syncthreads();               // done reading 'cur' before it is re-staged
    cur ^= 1;
  }

  const int hi = lane >> 4;
  const int n0 = lane & 15;
#pragma unroll
  for (int mt = 0; mt < 4; ++mt) {
#pragma unroll
    for (int nt = 0; nt < 2; ++nt) {
      const int col = blockN + waveN * 32 + nt * 16 + n0;
      if constexpr (OUTMODE == 2) {
        // transposed bf16: rows of C become the fast axis -> one b128 store
        const long row0 = blockM + waveM * 64 + mt * 16 + 8 * hi;  // 8-aligned
        const int bIdx = (int)(row0 >> 11);       // row / 2048
        const int s0   = (int)(row0 & 2047);
        Pack8 pk;
#pragma unroll
        for (int r = 0; r < 8; ++r) pk.h[r] = (bf16_t)acc[mt][nt][r];
        *(uint4*)((bf16_t*)Cout + ((long)bIdx * N + col) * 2048 + s0) = pk.q;
      } else {
#pragma unroll
        for (int r = 0; r < 8; ++r) {
          long row = blockM + waveM * 64 + mt * 16 + r + 8 * hi;
          if constexpr (OUTMODE == 1)
            ((bf16_t*)Cout)[row * N + col] = (bf16_t)acc[mt][nt][r];
          else
            ((float*)Cout)[row * N + col]  = acc[mt][nt][r];
        }
      }
    }
  }
}

// -------------------------------------------------------------------------
// RoPE in place on bf16 Q and K, layout [B*S, D], head h at cols h*64..h*64+63.
// Q is pre-scaled by (1/sqrt(dh)) * log2(e) so the softmax can run in exp2.
__global__ void rope_kernel(bf16_t* __restrict__ Q, bf16_t* __restrict__ Kt,
                            const int* __restrict__ pos) {
  const int S = 2048, D = 1024;
  const float QSCALE = 0.125f * 1.4426950408889634f;   // 1/sqrt(64) * log2(e)
  long idx = (long)blockIdx.x * blockDim.x + threadIdx.x;
  const long total = (long)4 * S * (D / 2);
  if (idx >= total) return;
  long rowp = idx >> 9;            // D/2 = 512 pairs per row
  int p = (int)(idx & 511);
  int s = (int)(rowp & (S - 1));
  int i = p & 31;                  // pair index within head (dh/2 = 32)
  int h = p >> 5;
  // theta^(-2i/64) = 2^(-i * log2(10000)/32)
  float inv = exp2f(-(float)i * 0.41524101186092029f);
  float ang = (float)pos[s] * inv;
  float sn, cs;
  __sincosf(ang, &sn, &cs);
  long base = rowp * D + (long)h * 64 + 2 * i;
  float q1 = (float)Q[base], q2 = (float)Q[base + 1];
  Q[base]     = (bf16_t)((q1 * cs - q2 * sn) * QSCALE);
  Q[base + 1] = (bf16_t)((q1 * sn + q2 * cs) * QSCALE);
  float k1 = (float)Kt[base], k2 = (float)Kt[base + 1];
  Kt[base]     = (bf16_t)(k1 * cs - k2 * sn);
  Kt[base + 1] = (bf16_t)(k1 * sn + k2 * cs);
}

// -------------------------------------------------------------------------
// One 32-key block of causal flash attention for one wave's 16-row Q tile.
// Scores are in log2 units (scale folded into Q); softmax uses exp2.
// MASKED=false: block is entirely below the diagonal -> no compares at all.
template<bool MASKED>
__device__ __forceinline__ void attn_block(
    int kb, int qs, int hi, int n0,
    const bf16_t* khead, const bf16_t* vhead, bf16_t* pb,
    v16bf qa0, v16bf qa1,
    v8f (&accv)[4], float (&m)[8], float (&l)[8])
{
  const int D = 1024, S = 2048;
  const v8f vzero = {0.f,0.f,0.f,0.f,0.f,0.f,0.f,0.f};

  // Issue ALL fragment loads up front so V-load latency hides behind
  // the score WMMAs and the softmax.
  const bf16_t* kb0 = khead + (long)kb * D;
  const bf16_t* kb1 = khead + (long)(kb + 16) * D;
  v16bf kf00 = load_frag_rm(kb0, D);
  v16bf kf01 = load_frag_rm(kb0 + 32, D);
  v16bf kf10 = load_frag_rm(kb1, D);
  v16bf kf11 = load_frag_rm(kb1 + 32, D);
  v16bf vf0 = load_frag_rm(vhead + (long)0  * S + kb, S);
  v16bf vf1 = load_frag_rm(vhead + (long)16 * S + kb, S);
  v16bf vf2 = load_frag_rm(vhead + (long)32 * S + kb, S);
  v16bf vf3 = load_frag_rm(vhead + (long)48 * S + kb, S);

  v8f c0 = wmma_bf16(qa0, kf00, vzero);
  c0 = wmma_bf16(qa1, kf01, c0);
  v8f c1 = wmma_bf16(qa0, kf10, vzero);
  c1 = wmma_bf16(qa1, kf11, c1);

  // C layout: row = qs + r + 8*hi, col = kb + 16j + n0
  float sv[2][8];
#pragma unroll
  for (int r = 0; r < 8; ++r) {
    if constexpr (MASKED) {
      const int row = qs + r + 8 * hi;
      sv[0][r] = (kb + n0      <= row) ? c0[r] : -__builtin_inff();
      sv[1][r] = (kb + 16 + n0 <= row) ? c1[r] : -__builtin_inff();
    } else {
      sv[0][r] = c0[r];
      sv[1][r] = c1[r];
    }
  }

  // online softmax in log2 domain; fused-DPP butterflies for row reductions
  float mnew[8], alpha[8];
#pragma unroll
  for (int r = 0; r < 8; ++r) {
    float t = rowmax16(fmax_raw(sv[0][r], sv[1][r]));
    mnew[r]  = fmax_raw(m[r], t);
    alpha[r] = exp2f(m[r] - mnew[r]);
    m[r] = mnew[r];
  }
#pragma unroll
  for (int r = 0; r < 8; ++r) {
    sv[0][r] = exp2f(sv[0][r] - mnew[r]);
    sv[1][r] = exp2f(sv[1][r] - mnew[r]);
    float rs = rowsum16(sv[0][r] + sv[1][r]);
    l[r] = l[r] * alpha[r] + rs;
    accv[0][r] *= alpha[r];
    accv[1][r] *= alpha[r];
    accv[2][r] *= alpha[r];
    accv[3][r] *= alpha[r];
  }

  // C-layout P -> LDS -> A-fragment layout
#pragma unroll
  for (int j = 0; j < 2; ++j)
#pragma unroll
    for (int r = 0; r < 8; ++r)
      pb[(r + 8 * hi) * 40 + 16 * j + n0] = (bf16_t)sv[j][r];

  v16bf pa = load_frag_rm(pb, 40);

  accv[0] = wmma_bf16(pa, vf0, accv[0]);
  accv[1] = wmma_bf16(pa, vf1, accv[1]);
  accv[2] = wmma_bf16(pa, vf2, accv[2]);
  accv[3] = wmma_bf16(pa, vf3, accv[3]);
}

// Causal flash attention. One wave per 16-row Q tile; 4 waves per block.
// V is pre-transposed: Vt[((b*1024 + h*64 + d))*2048 + s].
__global__ __launch_bounds__(128) void flash_attn_kernel(
    const bf16_t* __restrict__ Q, const bf16_t* __restrict__ K,
    const bf16_t* __restrict__ Vt, bf16_t* __restrict__ O)
{
  const int S = 2048, D = 1024;
  __shared__ bf16_t pbuf[4][16][40];      // per-wave P staging (C -> A layout)

  const int lane = threadIdx.x & 31;
  const int wid  = threadIdx.x >> 5;
  const int tile = blockIdx.x;            // B*H*(S/64) = 2048
  const int qblk = tile & 31;             // S/64 = 32
  const int bh   = tile >> 5;
  const int h    = bh & 15;
  const int b    = bh >> 4;
  const int qs   = qblk * 64 + wid * 16;
  const int hi   = lane >> 4;
  const int n0   = lane & 15;

  const bf16_t* khead = K + (long)b * S * D + h * 64;
  const bf16_t* vhead = Vt + ((long)b * D + h * 64) * S;   // per-head [64][S]
  bf16_t* pb = &pbuf[wid][0][0];

  const long qbase = (long)(b * S + qs) * D + h * 64;
  v16bf qa0 = load_frag_rm(Q + qbase, D);        // dh 0..31 (pre-scaled)
  v16bf qa1 = load_frag_rm(Q + qbase + 32, D);   // dh 32..63

  const v8f vzero = {0.f,0.f,0.f,0.f,0.f,0.f,0.f,0.f};
  v8f accv[4] = {vzero, vzero, vzero, vzero};
  float m[8], l[8];
#pragma unroll
  for (int r = 0; r < 8; ++r) { m[r] = -__builtin_inff(); l[r] = 0.f; }

  // Blocks with kb+31 <= qs need no causal mask; exactly one partial block.
  const int kfull = ((qs + 1) >> 5) << 5;
  for (int kb = 0; kb < kfull; kb += 32) {
    __builtin_prefetch(khead + (long)(kb + 32) * D, 0, 0);
    attn_block<false>(kb, qs, hi, n0, khead, vhead, pb, qa0, qa1, accv, m, l);
  }
  attn_block<true>(kfull, qs, hi, n0, khead, vhead, pb, qa0, qa1, accv, m, l);

  // normalize and write O (bf16, [B*S, D], head-major columns)
#pragma unroll
  for (int t4 = 0; t4 < 4; ++t4)
#pragma unroll
    for (int r = 0; r < 8; ++r) {
      float val = accv[t4][r] * __builtin_amdgcn_rcpf(l[r]);
      long orow = (long)(b * S + qs + r + 8 * hi) * D + h * 64 + 16 * t4 + n0;
      O[orow] = (bf16_t)val;
    }
}

// -------------------------------------------------------------------------
extern "C" void kernel_launch(void* const* d_in, const int* in_sizes, int n_in,
                              void* d_out, int out_size, void* d_ws, size_t ws_size,
                              hipStream_t stream) {
  const int Bb = 4, S = 2048, D = 1024;
  const long MS = (long)Bb * S;             // 8192 rows

  const float* X  = (const float*)d_in[0];
  const int*  pos = (const int*)d_in[1];
  const float* wq = (const float*)d_in[2];
  const float* wk = (const float*)d_in[3];
  const float* wv = (const float*)d_in[4];
  const float* wo = (const float*)d_in[5];

  char* ws = (char*)d_ws;
  bf16_t* Xb  = (bf16_t*)(ws);                    // 16 MiB
  bf16_t* Wqb = (bf16_t*)(ws + (16l << 20));      //  2 MiB each
  bf16_t* Wkb = (bf16_t*)(ws + (18l << 20));
  bf16_t* Wvb = (bf16_t*)(ws + (20l << 20));
  bf16_t* Wob = (bf16_t*)(ws + (22l << 20));
  bf16_t* Qb  = (bf16_t*)(ws + (24l << 20));      // 16 MiB each
  bf16_t* Kb  = (bf16_t*)(ws + (40l << 20));
  bf16_t* Vt  = (bf16_t*)(ws + (56l << 20));      // transposed V
  bf16_t* Ob  = (bf16_t*)(ws + (72l << 20));      // total 88 MiB

  const long nX = MS * D;                   // 8,388,608
  const long nW = (long)D * D;              // 1,048,576
  f32_to_bf16_kernel<<<(int)((nX + 255) / 256), 256, 0, stream>>>(X,  Xb,  nX);
  f32_to_bf16_kernel<<<(int)((nW + 255) / 256), 256, 0, stream>>>(wq, Wqb, nW);
  f32_to_bf16_kernel<<<(int)((nW + 255) / 256), 256, 0, stream>>>(wk, Wkb, nW);
  f32_to_bf16_kernel<<<(int)((nW + 255) / 256), 256, 0, stream>>>(wv, Wvb, nW);
  f32_to_bf16_kernel<<<(int)((nW + 255) / 256), 256, 0, stream>>>(wo, Wob, nW);

  dim3 ggrid(D / 128, (int)(MS / 128));     // (8, 64)
  gemm_nt_kernel<1><<<ggrid, 256, 0, stream>>>(Xb, Wqb, Qb, (int)MS, D, D);
  gemm_nt_kernel<1><<<ggrid, 256, 0, stream>>>(Xb, Wkb, Kb, (int)MS, D, D);
  gemm_nt_kernel<2><<<ggrid, 256, 0, stream>>>(Xb, Wvb, Vt, (int)MS, D, D);

  const long nPairs = MS * (D / 2);         // 4,194,304
  rope_kernel<<<(int)((nPairs + 255) / 256), 256, 0, stream>>>(Qb, Kb, pos);

  flash_attn_kernel<<<Bb * 16 * (S / 64), 128, 0, stream>>>(Qb, Kb, Vt, Ob);

  gemm_nt_kernel<0><<<ggrid, 256, 0, stream>>>(Ob, Wob, d_out, (int)MS, D, D);
}
